// TransformerEncoder_56066503082214
// MI455X (gfx1250) — matrix operand order repaired and checked by reading, not compiled, not run on hardware
//
#include <hip/hip_runtime.h>
#include <math.h>

// ---------------------------------------------------------------------------
// Problem constants (match reference)
// ---------------------------------------------------------------------------
#define BB   2
#define TT   16
#define NPAT 256
#define NLAT 256
#define NN   512            // NPAT + NLAT tokens per frame
#define DD   768
#define HH   12
#define HD   64
#define FFN  3072
#define NTOK (BB*TT*NN)     // 16384 tokens
#define NLAYERS 6
#define SOFTCAP 50.0f
#define EPSV 1e-6f

typedef __attribute__((ext_vector_type(16))) _Float16 v16h;
typedef __attribute__((ext_vector_type(8)))  _Float16 v8h;
typedef __attribute__((ext_vector_type(4)))  _Float16 v4h;
typedef __attribute__((ext_vector_type(8)))  float    v8f;

union F16Frag { v16h v; v8h h[2]; };

// ---------------------------------------------------------------------------
// Build X = concat(patches, broadcast latent) : (B,T,N,D) flat
// ---------------------------------------------------------------------------
__global__ __launch_bounds__(256) void build_x_kernel(
    const float* __restrict__ patches, const float* __restrict__ latent,
    float* __restrict__ X)
{
    long idx = (long)blockIdx.x * 256 + threadIdx.x;
    if (idx >= (long)NTOK * DD) return;
    long tok = idx / DD;  int d = (int)(idx % DD);
    int n = (int)(tok % NN); long bt = tok / NN;
    if (n < NPAT) X[idx] = patches[(bt * NPAT + n) * (long)DD + d];
    else          X[idx] = latent[(long)(n - NPAT) * DD + d];
}

// ---------------------------------------------------------------------------
// RoPE tables. Spatial 2D (NPAT x HD), Temporal 1D (TT x HD)
// ---------------------------------------------------------------------------
__global__ __launch_bounds__(256) void rope_spatial_kernel(float* c, float* s)
{
    int idx = blockIdx.x * 256 + threadIdx.x;   // NPAT*HD = 16384
    if (idx >= NPAT * HD) return;
    int n = idx / HD, d = idx % HD;
    int j = d & 31;                 // ang index (cos/sin duplicated across halves)
    int i = j & 15;                 // quarter index
    float pos = (j < 16) ? (float)(n / 16) : (float)(n % 16); // pos_h : pos_w
    float omega = powf(10000.f, -(float)i / 16.f);
    float ang = pos * omega;
    c[idx] = cosf(ang);  s[idx] = sinf(ang);
}

__global__ __launch_bounds__(256) void rope_temporal_kernel(float* c, float* s)
{
    int idx = blockIdx.x * 256 + threadIdx.x;   // TT*HD = 1024
    if (idx >= TT * HD) return;
    int t = idx / HD, d = idx % HD;
    int j = d & 31;
    float omega = powf(10000.f, -(float)j / 32.f);
    float ang = (float)t * omega;
    c[idx] = cosf(ang);  s[idx] = sinf(ang);
}

// ---------------------------------------------------------------------------
// RMSNorm: one wave per token row of 768
// ---------------------------------------------------------------------------
__global__ __launch_bounds__(256) void rmsnorm_kernel(
    const float* __restrict__ X, const float* __restrict__ w,
    float* __restrict__ XN)
{
    int wave = threadIdx.x >> 5, lane = threadIdx.x & 31;
    long tok = (long)blockIdx.x * 8 + wave;
    const float* x = X + tok * DD;
    float vals[DD / 32];
    float ss = 0.f;
#pragma unroll
    for (int i = 0; i < DD / 32; ++i) {
        float v = x[lane + i * 32];
        vals[i] = v;  ss += v * v;
    }
#pragma unroll
    for (int o = 16; o > 0; o >>= 1) ss += __shfl_xor(ss, o, 32);
    float r = rsqrtf(ss * (1.f / DD) + EPSV);
    float* y = XN + tok * DD;
#pragma unroll
    for (int i = 0; i < DD / 32; ++i) {
        int d = lane + i * 32;
        y[d] = vals[i] * r * w[d];
    }
}

// ---------------------------------------------------------------------------
// GEMM: C[M,N] = A[M,K] @ B[K,N] (+R), fp32 in/out, f16 WMMA, f32 accum.
// Block tile 128x128, BK=32, 256 threads (8 waves, each 64x32 -> 8 WMMA/step).
// Global->register double buffering hides L2 latency behind WMMA compute.
// EPI: 0 = plain store, 1 = C = acc + R (residual)
// ---------------------------------------------------------------------------
template <int EPI>
__global__ __launch_bounds__(256) void gemm_wmma_kernel(
    const float* __restrict__ A, const float* __restrict__ B,
    const float* __restrict__ R, float* __restrict__ C,
    int M, int N, int K)
{
    __shared__ _Float16 Al[128 * 32];     // Al[m*32+k]
    __shared__ _Float16 Bl[128 * 32];     // transposed: Bl[n*32+k]
    const int m0 = blockIdx.x * 128;
    const int n0 = blockIdx.y * 128;
    const int t = threadIdx.x;
    const int lane = t & 31;
    const int wave = t >> 5;
    const int wm = wave >> 2;             // 0..1
    const int wn = wave & 3;              // 0..3

    const v8f vzero = {0.f,0.f,0.f,0.f,0.f,0.f,0.f,0.f};
    v8f acc[4][2];
#pragma unroll
    for (int mi = 0; mi < 4; ++mi)
#pragma unroll
        for (int ni = 0; ni < 2; ++ni) acc[mi][ni] = vzero;

    const int kbA = (lane < 16) ? 0 : 8;
    const int kbB = (lane < 16) ? 0 : 8;

    float4 pa[4], pb[4];
    auto fetch = [&](int k0) {
#pragma unroll
        for (int i = 0; i < 4; ++i) {
            int idx = t + i * 256;
            int row = idx >> 3;                    // A: 0..127
            int c4  = (idx & 7) << 2;
            pa[i] = *(const float4*)(A + (size_t)(m0 + row) * K + k0 + c4);
            int kk = idx >> 5;                     // B: 0..31
            int n4 = (idx & 31) << 2;
            pb[i] = *(const float4*)(B + (size_t)(k0 + kk) * N + n0 + n4);
        }
    };
    auto stage = [&]() {
#pragma unroll
        for (int i = 0; i < 4; ++i) {
            int idx = t + i * 256;
            int row = idx >> 3;
            int c4  = (idx & 7) << 2;
            v4h hv = { (_Float16)pa[i].x, (_Float16)pa[i].y,
                       (_Float16)pa[i].z, (_Float16)pa[i].w };
            *(v4h*)(Al + row * 32 + c4) = hv;
            int kk = idx >> 5;
            int n4 = (idx & 31) << 2;
            Bl[(n4 + 0) * 32 + kk] = (_Float16)pb[i].x;
            Bl[(n4 + 1) * 32 + kk] = (_Float16)pb[i].y;
            Bl[(n4 + 2) * 32 + kk] = (_Float16)pb[i].z;
            Bl[(n4 + 3) * 32 + kk] = (_Float16)pb[i].w;
        }
    };

    fetch(0);
    for (int k0 = 0; k0 < K; k0 += 32) {
        stage();
        __syncthreads();
        if (k0 + 32 < K) fetch(k0 + 32);          // prefetch while WMMAs run

        F16Frag bfr[2];
#pragma unroll
        for (int ni = 0; ni < 2; ++ni) {
            int n = wn * 32 + ni * 16 + (lane & 15);
            bfr[ni].h[0] = *(const v8h*)(Bl + n * 32 + kbB);
            bfr[ni].h[1] = *(const v8h*)(Bl + n * 32 + kbB + 16);
        }
#pragma unroll
        for (int mi = 0; mi < 4; ++mi) {
            int m = wm * 64 + mi * 16 + (lane & 15);
            F16Frag afr;
            afr.h[0] = *(const v8h*)(Al + m * 32 + kbA);
            afr.h[1] = *(const v8h*)(Al + m * 32 + kbA + 16);
#pragma unroll
            for (int ni = 0; ni < 2; ++ni)
                acc[mi][ni] = __builtin_amdgcn_wmma_f32_16x16x32_f16(
                    false, afr.v, false, bfr[ni].v, (short)0, acc[mi][ni],
                    false, false);
        }
        __syncthreads();
    }

    const int rr = (lane < 16) ? 0 : 8;
    const int cc = lane & 15;
#pragma unroll
    for (int mi = 0; mi < 4; ++mi)
#pragma unroll
        for (int ni = 0; ni < 2; ++ni) {
            int baseM = m0 + wm * 64 + mi * 16 + rr;
            int baseN = n0 + wn * 32 + ni * 16 + cc;
#pragma unroll
            for (int r = 0; r < 8; ++r) {
                size_t off = (size_t)(baseM + r) * N + baseN;
                float v = acc[mi][ni][r];
                if (EPI == 1) v += R[off];
                C[off] = v;
            }
        }
}

// ---------------------------------------------------------------------------
// Dual GEMM + SwiGLU gate: G = silu(A@B1) * (A@B2).
// Block tile 128x64, 8 waves each own a 64x16 strip (8 WMMA/step total),
// with the same global->register double buffering.
// ---------------------------------------------------------------------------
__global__ __launch_bounds__(256) void gemm_swiglu_kernel(
    const float* __restrict__ A, const float* __restrict__ B1,
    const float* __restrict__ B2, float* __restrict__ G,
    int M, int N, int K)
{
    __shared__ _Float16 Al[128 * 32];
    __shared__ _Float16 B1l[64 * 32];     // transposed
    __shared__ _Float16 B2l[64 * 32];     // transposed
    const int m0 = blockIdx.x * 128;
    const int n0 = blockIdx.y * 64;
    const int t = threadIdx.x;
    const int lane = t & 31;
    const int wave = t >> 5;
    const int wm = wave >> 2;             // 0..1
    const int wn = wave & 3;              // 0..3

    const v8f vzero = {0.f,0.f,0.f,0.f,0.f,0.f,0.f,0.f};
    v8f acc1[4], acc2[4];
#pragma unroll
    for (int mi = 0; mi < 4; ++mi) { acc1[mi] = vzero; acc2[mi] = vzero; }

    const int kbA = (lane < 16) ? 0 : 8;
    const int kbB = (lane < 16) ? 0 : 8;

    float4 pa[4], pb1[2], pb2[2];
    auto fetch = [&](int k0) {
#pragma unroll
        for (int i = 0; i < 4; ++i) {
            int idx = t + i * 256;
            int row = idx >> 3;
            int c4  = (idx & 7) << 2;
            pa[i] = *(const float4*)(A + (size_t)(m0 + row) * K + k0 + c4);
        }
#pragma unroll
        for (int i = 0; i < 2; ++i) {
            int idx = t + i * 256;                 // 0..511 float4 slots
            int kk = idx >> 4;                     // 0..31
            int n4 = (idx & 15) << 2;
            pb1[i] = *(const float4*)(B1 + (size_t)(k0 + kk) * N + n0 + n4);
            pb2[i] = *(const float4*)(B2 + (size_t)(k0 + kk) * N + n0 + n4);
        }
    };
    auto stage = [&]() {
#pragma unroll
        for (int i = 0; i < 4; ++i) {
            int idx = t + i * 256;
            int row = idx >> 3;
            int c4  = (idx & 7) << 2;
            v4h hv = { (_Float16)pa[i].x, (_Float16)pa[i].y,
                       (_Float16)pa[i].z, (_Float16)pa[i].w };
            *(v4h*)(Al + row * 32 + c4) = hv;
        }
#pragma unroll
        for (int i = 0; i < 2; ++i) {
            int idx = t + i * 256;
            int kk = idx >> 4;
            int n4 = (idx & 15) << 2;
            B1l[(n4 + 0) * 32 + kk] = (_Float16)pb1[i].x;
            B1l[(n4 + 1) * 32 + kk] = (_Float16)pb1[i].y;
            B1l[(n4 + 2) * 32 + kk] = (_Float16)pb1[i].z;
            B1l[(n4 + 3) * 32 + kk] = (_Float16)pb1[i].w;
            B2l[(n4 + 0) * 32 + kk] = (_Float16)pb2[i].x;
            B2l[(n4 + 1) * 32 + kk] = (_Float16)pb2[i].y;
            B2l[(n4 + 2) * 32 + kk] = (_Float16)pb2[i].z;
            B2l[(n4 + 3) * 32 + kk] = (_Float16)pb2[i].w;
        }
    };

    fetch(0);
    for (int k0 = 0; k0 < K; k0 += 32) {
        stage();
        __syncthreads();
        if (k0 + 32 < K) fetch(k0 + 32);

        F16Frag b1f, b2f;
        {
            int n = wn * 16 + (lane & 15);
            b1f.h[0] = *(const v8h*)(B1l + n * 32 + kbB);
            b1f.h[1] = *(const v8h*)(B1l + n * 32 + kbB + 16);
            b2f.h[0] = *(const v8h*)(B2l + n * 32 + kbB);
            b2f.h[1] = *(const v8h*)(B2l + n * 32 + kbB + 16);
        }
#pragma unroll
        for (int mi = 0; mi < 4; ++mi) {
            int m = wm * 64 + mi * 16 + (lane & 15);
            F16Frag afr;
            afr.h[0] = *(const v8h*)(Al + m * 32 + kbA);
            afr.h[1] = *(const v8h*)(Al + m * 32 + kbA + 16);
            acc1[mi] = __builtin_amdgcn_wmma_f32_16x16x32_f16(
                false, afr.v, false, b1f.v, (short)0, acc1[mi], false, false);
            acc2[mi] = __builtin_amdgcn_wmma_f32_16x16x32_f16(
                false, afr.v, false, b2f.v, (short)0, acc2[mi], false, false);
        }
        __syncthreads();
    }

    const int rr = (lane < 16) ? 0 : 8;
    const int cc = lane & 15;
#pragma unroll
    for (int mi = 0; mi < 4; ++mi) {
        int baseM = m0 + wm * 64 + mi * 16 + rr;
        int baseN = n0 + wn * 16 + cc;
#pragma unroll
        for (int r = 0; r < 8; ++r) {
            size_t off = (size_t)(baseM + r) * N + baseN;
            float h1 = acc1[mi][r];
            float h2 = acc2[mi][r];
            float g  = h1 / (1.f + expf(-h1)) * h2;   // silu(h1)*h2
            G[off] = g;
        }
    }
}

// ---------------------------------------------------------------------------
// QKNorm (RMS over head dim, learnable gain) + RoPE.
// One wave per (token, head); lane owns dims d=2*lane, 2*lane+1.
// mode 0: spatial (pos = n, rope only for n < NPAT); mode 1: temporal (pos=t)
// ---------------------------------------------------------------------------
__global__ __launch_bounds__(256) void qknorm_rope_kernel(
    float* __restrict__ Q, float* __restrict__ Kv,
    const float* __restrict__ qg, const float* __restrict__ kg,
    const float* __restrict__ ct, const float* __restrict__ st, int mode)
{
    int wave = threadIdx.x >> 5, lane = threadIdx.x & 31;
    long rid = (long)blockIdx.x * 8 + wave;     // token*HH + h
    int h = (int)(rid % HH);
    long tok = rid / HH;
    int d0 = h * HD;
    int d = lane * 2;

    int pos, doRope;
    if (mode == 0) { int n = (int)(tok % NN); doRope = (n < NPAT); pos = doRope ? n : 0; }
    else           { pos = (int)((tok / NN) % TT); doRope = 1; }

    float c0 = 1.f, c1 = 1.f, s0 = 0.f, s1 = 0.f;
    if (doRope) {
        c0 = ct[pos * HD + d];     c1 = ct[pos * HD + d + 1];
        s0 = st[pos * HD + d];     s1 = st[pos * HD + d + 1];
    }

#pragma unroll
    for (int which = 0; which < 2; ++which) {
        float* P = which ? Kv : Q;
        const float* g = which ? kg : qg;
        size_t off = (size_t)tok * DD + d0 + d;
        float x0 = P[off], x1 = P[off + 1];
        float ss = x0 * x0 + x1 * x1;
#pragma unroll
        for (int o = 16; o > 0; o >>= 1) ss += __shfl_xor(ss, o, 32);
        float r = rsqrtf(ss * (1.f / HD) + EPSV);
        x0 = x0 * r * g[d];
        x1 = x1 * r * g[d + 1];
        // rot_half partner lives in lane^16 (dim d^32)
        float p0 = __shfl_xor(x0, 16, 32);
        float p1 = __shfl_xor(x1, 16, 32);
        if (doRope) {
            float rh0 = (lane < 16) ? -p0 : p0;
            float rh1 = (lane < 16) ? -p1 : p1;
            x0 = x0 * c0 + rh0 * s0;
            x1 = x1 * c1 + rh1 * s1;
        }
        P[off] = x0;  P[off + 1] = x1;
    }
}

// ---------------------------------------------------------------------------
// Attention: one wave per (sequence, head, query). fp32 exact path.
// MODE 0: spatial, seq=b*T+t, S=NN, no mask. MODE 1: temporal, seq=b*NN+n,
// S=TT, causal. Scores softcapped with 50*tanh(s/50) before mask/softmax.
// ---------------------------------------------------------------------------
template <int S, bool CAUSAL, int MODE>
__global__ __launch_bounds__(256) void attention_kernel(
    const float* __restrict__ Q, const float* __restrict__ Kv,
    const float* __restrict__ V, float* __restrict__ O)
{
    __shared__ float sc[8][S];
    __shared__ float qs[8][HD];
    int wave = threadIdx.x >> 5, lane = threadIdx.x & 31;
    long rid = (long)blockIdx.x * 8 + wave;
    int qp  = (int)(rid % S);
    int h   = (int)((rid / S) % HH);
    int seq = (int)(rid / ((long)S * HH));
    int d0  = h * HD;

    auto tokOf = [&](int p) -> long {
        if (MODE == 0) return (long)seq * NN + p;
        else           return (long)(seq / NN) * (TT * NN) + (long)p * NN + (seq % NN);
    };
    long qtok = tokOf(qp);

    qs[wave][lane * 2]     = Q[qtok * DD + d0 + lane * 2];
    qs[wave][lane * 2 + 1] = Q[qtok * DD + d0 + lane * 2 + 1];
    __syncthreads();

    const float scale = rsqrtf((float)HD);
    float lmax = -INFINITY;
    for (int jj = 0; jj < S; jj += 32) {
        int j = jj + lane;
        if (j < S) {
            const float* kr = Kv + tokOf(j) * DD + d0;
            float acc = 0.f;
#pragma unroll
            for (int d4 = 0; d4 < HD; d4 += 4) {
                float4 kx = *(const float4*)(kr + d4);
                float4 qx = *(const float4*)(&qs[wave][d4]);
                acc += qx.x * kx.x + qx.y * kx.y + qx.z * kx.z + qx.w * kx.w;
            }
            float s = acc * scale;
            s = SOFTCAP * tanhf(s * (1.f / SOFTCAP));
            if (CAUSAL && j > qp) s = -INFINITY;
            sc[wave][j] = s;
            lmax = fmaxf(lmax, s);
        }
    }
#pragma unroll
    for (int o = 16; o > 0; o >>= 1) lmax = fmaxf(lmax, __shfl_xor(lmax, o, 32));

    float lsum = 0.f;
    for (int jj = 0; jj < S; jj += 32) {
        int j = jj + lane;
        if (j < S) {
            float e = expf(sc[wave][j] - lmax);
            sc[wave][j] = e;
            lsum += e;
        }
    }
#pragma unroll
    for (int o = 16; o > 0; o >>= 1) lsum += __shfl_xor(lsum, o, 32);
    float inv = 1.f / lsum;
    __syncthreads();

    float o0 = 0.f, o1 = 0.f;
    for (int j = 0; j < S; ++j) {
        float a = sc[wave][j];
        const float* vr = V + tokOf(j) * DD + d0 + lane * 2;
        float2 vv = *(const float2*)vr;
        o0 += a * vv.x;
        o1 += a * vv.y;
    }
    O[qtok * DD + d0 + lane * 2]     = o0 * inv;
    O[qtok * DD + d0 + lane * 2 + 1] = o1 * inv;
}

// ---------------------------------------------------------------------------
// Output: latent tokens only, (B,T,NLAT,D)
// ---------------------------------------------------------------------------
__global__ __launch_bounds__(256) void copy_out_kernel(
    const float* __restrict__ X, float* __restrict__ out)
{
    long idx = (long)blockIdx.x * 256 + threadIdx.x;   // BB*TT*NLAT*DD
    if (idx >= (long)BB * TT * NLAT * DD) return;
    long tok = idx / DD;  int d = (int)(idx % DD);
    long bt = tok / NLAT; int nl = (int)(tok % NLAT);
    out[idx] = X[((bt * NN) + NPAT + nl) * (long)DD + d];
}

// ---------------------------------------------------------------------------
// Host driver
// ---------------------------------------------------------------------------
extern "C" void kernel_launch(void* const* d_in, const int* in_sizes, int n_in,
                              void* d_out, int out_size, void* d_ws, size_t ws_size,
                              hipStream_t stream)
{
    const float* patches = (const float*)d_in[0];
    const float* latent  = (const float*)d_in[1];

    // Workspace layout (floats): X | XN | S1 (Q|K|V|O, later G) | rope tables
    float* X  = (float*)d_ws;
    float* XN = X  + (size_t)NTOK * DD;
    float* S1 = XN + (size_t)NTOK * DD;
    float* Qb = S1;
    float* Kb = S1 + (size_t)NTOK * DD;
    float* Vb = S1 + (size_t)2 * NTOK * DD;
    float* Ob = S1 + (size_t)3 * NTOK * DD;
    float* Gb = S1;                              // aliases QKVO after attention
    float* cS = S1 + (size_t)4 * NTOK * DD;      // NPAT*HD
    float* sS = cS + NPAT * HD;
    float* cT = sS + NPAT * HD;                  // TT*HD
    float* sT = cT + TT * HD;

    build_x_kernel<<<(NTOK * DD + 255) / 256, 256, 0, stream>>>(patches, latent, X);
    rope_spatial_kernel<<<(NPAT * HD + 255) / 256, 256, 0, stream>>>(cS, sS);
    rope_temporal_kernel<<<(TT * HD + 255) / 256, 256, 0, stream>>>(cT, sT);

    const dim3 gProj(NTOK / 128, DD / 128);       // 128 x 6
    const dim3 gFfn (NTOK / 128, FFN / 64);       // 128 x 48
    const int  gNorm = NTOK / 8;                  // 2048
    const int  gRows = (NTOK * HH) / 8;           // 24576 (qknorm + attention)

    for (int l = 0; l < NLAYERS; ++l) {
        const int base = 2 + l * 11;              // insertion order within layer dict
        const float* norm1 = (const float*)d_in[base + 0];
        const float* norm2 = (const float*)d_in[base + 1];
        const float* qg    = (const float*)d_in[base + 2];
        const float* kg    = (const float*)d_in[base + 3];
        const float* wq    = (const float*)d_in[base + 4];
        const float* wk    = (const float*)d_in[base + 5];
        const float* wv    = (const float*)d_in[base + 6];
        const float* wo    = (const float*)d_in[base + 7];
        const float* w1    = (const float*)d_in[base + 8];
        const float* w2    = (const float*)d_in[base + 9];
        const float* w3    = (const float*)d_in[base + 10];
        const int mode = l & 1;                   // 0 spatial, 1 temporal

        // ---- attention block ----
        rmsnorm_kernel<<<gNorm, 256, 0, stream>>>(X, norm1, XN);
        gemm_wmma_kernel<0><<<gProj, 256, 0, stream>>>(XN, wq, nullptr, Qb, NTOK, DD, DD);
        gemm_wmma_kernel<0><<<gProj, 256, 0, stream>>>(XN, wk, nullptr, Kb, NTOK, DD, DD);
        gemm_wmma_kernel<0><<<gProj, 256, 0, stream>>>(XN, wv, nullptr, Vb, NTOK, DD, DD);
        qknorm_rope_kernel<<<gRows, 256, 0, stream>>>(Qb, Kb, qg, kg,
                                                      mode ? cT : cS, mode ? sT : sS, mode);
        if (mode == 0)
            attention_kernel<NN, false, 0><<<gRows, 256, 0, stream>>>(Qb, Kb, Vb, Ob);
        else
            attention_kernel<TT, true, 1><<<gRows, 256, 0, stream>>>(Qb, Kb, Vb, Ob);
        gemm_wmma_kernel<1><<<gProj, 256, 0, stream>>>(Ob, wo, X, X, NTOK, DD, DD);

        // ---- SwiGLU FFN ----
        rmsnorm_kernel<<<gNorm, 256, 0, stream>>>(X, norm2, XN);
        gemm_swiglu_kernel<<<gFfn, 256, 0, stream>>>(XN, w1, w2, Gb, NTOK, FFN, DD);
        gemm_wmma_kernel<1><<<gProj, 256, 0, stream>>>(Gb, w3, X, X, NTOK, DD, FFN);
    }

    copy_out_kernel<<<((long)BB * TT * NLAT * DD + 255) / 256, 256, 0, stream>>>(
        X, (float*)d_out);
}
